// PGSA_11098195493008
// MI455X (gfx1250) — compile-verified, matrix operand order checked by reading, not run on hardware
//
#include <hip/hip_runtime.h>
#include <hip/hip_bf16.h>
#include <math.h>

// ---------------------------------------------------------------------------
// PGSA block for MI455X (gfx1250).  Heavy GEMMs / convs use
// v_wmma_f32_16x16x32_bf16 (bf16 in, fp32 accumulate); weight tiles are staged
// into LDS with GLOBAL_LOAD_ASYNC_TO_LDS_B128 (ASYNCcnt + s_wait_asynccnt).
// Shapes fixed: B=16, H=W=64, N=4096, C=256, M=B*N=65536.
// ---------------------------------------------------------------------------

#define Bsz   16
#define Npix  4096
#define Cch   256
#define Mrow  65536   // B*N

typedef __attribute__((ext_vector_type(16))) __bf16 v16bf;
typedef __attribute__((ext_vector_type(8)))  __bf16 v8bf;
typedef __attribute__((ext_vector_type(8)))  float  v8f;

// ---------------------------- async copy to LDS ----------------------------
// ISA 15.18.3: GLOBAL_LOAD_ASYNC_TO_LDS_B128 — VDST holds the LDS byte offset,
// VADDR the 64-bit global address.  Low 32 bits of a generic LDS pointer are
// the wave-relative LDS offset (flat aperture mapping), consistent with the
// compiler's own ds_load addressing of the same __shared__ array.
__device__ __forceinline__ void async_load_b128(const void* gsrc, void* ldst) {
  asm volatile("global_load_async_to_lds_b128 %0, %1, off"
               :: "v"((unsigned)(unsigned long long)ldst), "v"(gsrc)
               : "memory");
}
__device__ __forceinline__ void async_wait0() {
  asm volatile("s_wait_asynccnt 0x0" ::: "memory");
}

// ---------------------------- reductions -----------------------------------
__device__ __forceinline__ float waveRedSum(float v) {
#pragma unroll
  for (int o = 16; o > 0; o >>= 1) v += __shfl_xor(v, o, 32);
  return v;
}
__device__ __forceinline__ float waveRedMax(float v) {
#pragma unroll
  for (int o = 16; o > 0; o >>= 1) v = fmaxf(v, __shfl_xor(v, o, 32));
  return v;
}
__device__ __forceinline__ float waveRedMin(float v) {
#pragma unroll
  for (int o = 16; o > 0; o >>= 1) v = fminf(v, __shfl_xor(v, o, 32));
  return v;
}
__device__ __forceinline__ float blockRedSum(float v) {
  __shared__ float tmp[8];
  int lane = threadIdx.x & 31, wid = threadIdx.x >> 5;
  v = waveRedSum(v);
  if (lane == 0) tmp[wid] = v;
  __syncthreads();
  float r = (lane < 8) ? tmp[lane] : 0.0f;
  r = waveRedSum(r);
  __syncthreads();
  return r;
}
__device__ __forceinline__ float blockRedMax(float v) {
  __shared__ float tmp[8];
  int lane = threadIdx.x & 31, wid = threadIdx.x >> 5;
  v = waveRedMax(v);
  if (lane == 0) tmp[wid] = v;
  __syncthreads();
  float r = (lane < 8) ? tmp[lane] : -3.4e38f;
  r = waveRedMax(r);
  __syncthreads();
  return r;
}
__device__ __forceinline__ float blockRedMin(float v) {
  __shared__ float tmp[8];
  int lane = threadIdx.x & 31, wid = threadIdx.x >> 5;
  v = waveRedMin(v);
  if (lane == 0) tmp[wid] = v;
  __syncthreads();
  float r = (lane < 8) ? tmp[lane] : 3.4e38f;
  r = waveRedMin(r);
  __syncthreads();
  return r;
}

// ---------------------------- WMMA fragments -------------------------------
// 16x32 bf16 fragment: lane L<16 holds row L, k {0..7,16..23}; lane L>=16 holds
// row L-16, k {8..15,24..31}.  Per-lane pointer p = row_base + (lane>>4)*8.
__device__ __forceinline__ v16bf frag_from_bf(const __bf16* p) {
  v8bf lo = *(const v8bf*)p;
  v8bf hi = *(const v8bf*)(p + 16);
  v16bf f;
#pragma unroll
  for (int i = 0; i < 8; i++) { f[i] = lo[i]; f[i + 8] = hi[i]; }
  return f;
}
__device__ __forceinline__ v16bf frag_from_f32(const float* p) {
  v16bf f;
#pragma unroll
  for (int i = 0; i < 8; i++) f[i] = (__bf16)p[i];
#pragma unroll
  for (int i = 0; i < 8; i++) f[i + 8] = (__bf16)p[16 + i];
  return f;
}
__device__ __forceinline__ v16bf frag_zero() {
  v16bf f;
#pragma unroll
  for (int i = 0; i < 16; i++) f[i] = (__bf16)0.0f;
  return f;
}

// ---------------------------- generic GEMM ---------------------------------
// out[M,256] = epi( A[M,256] @ W^T ), W row-major (Cout=256, K=256) bf16.
// Block = 128x128 tile, 8 waves (4 along M x 2 along N), each wave 32x64.
// The block's 128x256 weight tile (64KB) is staged once into LDS via async
// copy; B fragments then come from ds_load_b128.
// ACT: 0 none, 1 relu, 2 gelu(exact).  HAS_BB: per-batch bias bb[b*256+col].
// HAS_CB: per-channel bias cb[col].  HAS_RES: += resid[row*256+col] (after act).
template <int ACT, bool HAS_BB, bool HAS_CB, bool HAS_RES, bool A_BF>
__global__ __launch_bounds__(256) void gemm256_kernel(
    const void* __restrict__ Ap, const __bf16* __restrict__ Wb,
    float* __restrict__ out, const float* __restrict__ bb,
    const float* __restrict__ cb, const float* __restrict__ resid) {
  __shared__ __bf16 ldsW[128 * 256];  // 64 KB weight tile

  const int tid  = threadIdx.x;
  const int lane = tid & 31;
  const int wid  = tid >> 5;
  const int mTile = blockIdx.x * 128 + (wid & 3) * 32;
  const int nLoc  = (wid >> 2) * 64;               // wave's col base within tile
  const int nTile = blockIdx.y * 128 + nLoc;       // global col base
  const int r = lane & 15, hlf = (lane >> 4) * 8;

  // ---- async stage of the weight tile ----
  {
    const __bf16* src = Wb + (size_t)blockIdx.y * 128 * 256;
#pragma unroll
    for (int j = 0; j < 16; j++) {
      int ch = tid + 256 * j;                      // 4096 x 16B chunks
      async_load_b128(src + ch * 8, &ldsW[ch * 8]);
    }
    async_wait0();
  }
  __syncthreads();

  v8f acc[2][4];
#pragma unroll
  for (int i = 0; i < 2; i++)
#pragma unroll
    for (int j = 0; j < 4; j++)
#pragma unroll
      for (int e = 0; e < 8; e++) acc[i][j][e] = 0.0f;

  for (int kk = 0; kk < 256; kk += 32) {
    v16bf aF[2], bF[4];
#pragma unroll
    for (int mi = 0; mi < 2; mi++) {
      size_t off = (size_t)(mTile + mi * 16 + r) * 256 + kk + hlf;
      if constexpr (A_BF) aF[mi] = frag_from_bf((const __bf16*)Ap + off);
      else                aF[mi] = frag_from_f32((const float*)Ap + off);
    }
#pragma unroll
    for (int ni = 0; ni < 4; ni++)
      bF[ni] = frag_from_bf(&ldsW[(nLoc + ni * 16 + r) * 256 + kk + hlf]);
#pragma unroll
    for (int mi = 0; mi < 2; mi++)
#pragma unroll
      for (int ni = 0; ni < 4; ni++)
        acc[mi][ni] = __builtin_amdgcn_wmma_f32_16x16x32_bf16(
            false, aF[mi], false, bF[ni], (short)0, acc[mi][ni], false, false);
  }

  const int colOff = lane & 15;
  const int rowHalf = (lane >> 4) * 8;
#pragma unroll
  for (int mi = 0; mi < 2; mi++)
#pragma unroll
    for (int ni = 0; ni < 4; ni++) {
      int col = nTile + ni * 16 + colOff;
#pragma unroll
      for (int rr = 0; rr < 8; rr++) {
        int row = mTile + mi * 16 + rowHalf + rr;
        float v = acc[mi][ni][rr];
        if constexpr (HAS_BB) v += bb[(row >> 12) * 256 + col];
        if constexpr (HAS_CB) v += cb[col];
        if constexpr (ACT == 1) v = fmaxf(v, 0.0f);
        if constexpr (ACT == 2) v = 0.5f * v * (1.0f + erff(v * 0.70710678118654752f));
        if constexpr (HAS_RES) v += resid[(size_t)row * 256 + col];
        out[(size_t)row * 256 + col] = v;
      }
    }
}

// ---------------------------- 3x3 conv (implicit GEMM) ---------------------
// Channels-last NHWC, zero padding 1.  Weights pre-arranged [tap][o][i] bf16.
// Each tap's 128x256 weight tile is async-staged into LDS.
// out = relu(conv(in)) (+ resid if HAS_RES, added AFTER relu).
template <bool HAS_RES>
__global__ __launch_bounds__(256) void conv3x3_kernel(
    const float* __restrict__ in, const __bf16* __restrict__ Wt,
    float* __restrict__ out, const float* __restrict__ resid) {
  __shared__ __bf16 ldsW[128 * 256];  // 64 KB per-tap weight tile

  const int tid  = threadIdx.x;
  const int lane = tid & 31;
  const int wid  = tid >> 5;
  const int mTile = blockIdx.x * 128 + (wid & 3) * 32;
  const int nLoc  = (wid >> 2) * 64;
  const int nTile = blockIdx.y * 128 + nLoc;
  const int r = lane & 15, hlf = (lane >> 4) * 8;

  int mRow[2], yv[2], xv[2];
#pragma unroll
  for (int mi = 0; mi < 2; mi++) {
    mRow[mi] = mTile + mi * 16 + r;
    yv[mi] = (mRow[mi] >> 6) & 63;
    xv[mi] = mRow[mi] & 63;
  }

  v8f acc[2][4];
#pragma unroll
  for (int i = 0; i < 2; i++)
#pragma unroll
    for (int j = 0; j < 4; j++)
#pragma unroll
      for (int e = 0; e < 8; e++) acc[i][j][e] = 0.0f;

  for (int t = 0; t < 9; t++) {
    __syncthreads();  // previous tap's LDS reads complete before overwrite
    {
      const __bf16* src = Wt + ((size_t)t * 256 + blockIdx.y * 128) * 256;
#pragma unroll
      for (int j = 0; j < 16; j++) {
        int ch = tid + 256 * j;
        async_load_b128(src + ch * 8, &ldsW[ch * 8]);
      }
      async_wait0();
    }
    __syncthreads();

    int dy = t / 3 - 1, dx = t % 3 - 1;
    bool val[2];
    const float* base[2];
#pragma unroll
    for (int mi = 0; mi < 2; mi++) {
      val[mi] = ((unsigned)(yv[mi] + dy) < 64u) && ((unsigned)(xv[mi] + dx) < 64u);
      base[mi] = in + ((long)mRow[mi] + (long)dy * 64 + dx) * 256;
    }
    for (int kk = 0; kk < 256; kk += 32) {
      v16bf aF[2], bF[4];
#pragma unroll
      for (int mi = 0; mi < 2; mi++)
        aF[mi] = val[mi] ? frag_from_f32(base[mi] + kk + hlf) : frag_zero();
#pragma unroll
      for (int ni = 0; ni < 4; ni++)
        bF[ni] = frag_from_bf(&ldsW[(nLoc + ni * 16 + r) * 256 + kk + hlf]);
#pragma unroll
      for (int mi = 0; mi < 2; mi++)
#pragma unroll
        for (int ni = 0; ni < 4; ni++)
          acc[mi][ni] = __builtin_amdgcn_wmma_f32_16x16x32_bf16(
              false, aF[mi], false, bF[ni], (short)0, acc[mi][ni], false, false);
    }
  }

  const int colOff = lane & 15;
  const int rowHalf = (lane >> 4) * 8;
#pragma unroll
  for (int mi = 0; mi < 2; mi++)
#pragma unroll
    for (int ni = 0; ni < 4; ni++) {
      int col = nTile + ni * 16 + colOff;
#pragma unroll
      for (int rr = 0; rr < 8; rr++) {
        int row = mTile + mi * 16 + rowHalf + rr;
        float v = fmaxf(acc[mi][ni][rr], 0.0f);
        if constexpr (HAS_RES) v += resid[(size_t)row * 256 + col];
        out[(size_t)row * 256 + col] = v;
      }
    }
}

// ---------------------------- small kernels --------------------------------
// LayerNorm over last dim (256), output bf16.  One wave per row.
__global__ __launch_bounds__(256) void ln_kernel(
    const float* __restrict__ x, const float* __restrict__ g,
    const float* __restrict__ b, __bf16* __restrict__ out) {
  int wid = threadIdx.x >> 5, lane = threadIdx.x & 31;
  int row = blockIdx.x * 8 + wid;
  const float* xr = x + (size_t)row * 256;
  float vals[8], s = 0.0f, s2 = 0.0f;
#pragma unroll
  for (int j = 0; j < 8; j++) {
    float t = xr[lane + 32 * j];
    vals[j] = t; s += t; s2 += t * t;
  }
  s = waveRedSum(s); s2 = waveRedSum(s2);
  float mean = s * (1.0f / 256.0f);
  float var  = s2 * (1.0f / 256.0f) - mean * mean;
  float rstd = rsqrtf(var + 1e-5f);
#pragma unroll
  for (int j = 0; j < 8; j++) {
    int c = lane + 32 * j;
    out[(size_t)row * 256 + c] = (__bf16)((vals[j] - mean) * rstd * g[c] + b[c]);
  }
}

__global__ void cvt_rows_kernel(const float* __restrict__ src, __bf16* __restrict__ dst,
                                int stride, int off) {
  int idx = blockIdx.x * 256 + threadIdx.x;  // 256*256 elements
  int rr = idx >> 8, c = idx & 255;
  dst[idx] = (__bf16)src[(size_t)rr * stride + off + c];
}

// OIHW (256,256,3,3) -> [tap][o][i] bf16
__global__ void conv_prep_kernel(const float* __restrict__ src, __bf16* __restrict__ dst) {
  int idx = blockIdx.x * 256 + threadIdx.x;  // 9*256*256
  int t = idx >> 16, o = (idx >> 8) & 255, i = idx & 255;
  dst[idx] = (__bf16)src[((size_t)o * 256 + i) * 9 + t];
}

__global__ void area_kernel(const float* __restrict__ mask, float* __restrict__ area, int inv) {
  int b = blockIdx.x;
  float s = 0.0f;
  for (int j = threadIdx.x; j < Npix; j += 256) {
    float m = mask[(size_t)b * Npix + j];
    s += inv ? (1.0f - m) : m;
  }
  s = blockRedSum(s);
  if (threadIdx.x == 0) area[b] = s;
}

// weighted GAP: out[b,c] = sum_n feat[b,n,c]*w(n) / (area[b]+5e-4)
__global__ void wgap_kernel(const float* __restrict__ feat, const float* __restrict__ mask,
                            int inv, const float* __restrict__ area, float* __restrict__ out) {
  int b = blockIdx.x, c = threadIdx.x;
  const float* fb = feat + (size_t)b * Npix * 256;
  const float* mb = mask + (size_t)b * Npix;
  float acc = 0.0f;
  for (int n = 0; n < Npix; n++) {
    float m = mb[n];
    if (inv) m = 1.0f - m;
    acc += fb[(size_t)n * 256 + c] * m;
  }
  out[b * 256 + c] = acc / (area[b] + 5e-4f);
}

// bias1[b,o] = sum_c fg[b,c]*(res1_w[o,256+c]+res1_w[o,512+c])
__global__ void bias1_kernel(const float* __restrict__ fg, const float* __restrict__ w,
                             float* __restrict__ out) {
  int b = blockIdx.x, o = threadIdx.x;
  const float* wr = w + (size_t)o * 768;
  const float* f = fg + b * 256;
  float acc = 0.0f;
  for (int c = 0; c < 256; c++) acc += f[c] * (wr[256 + c] + wr[512 + c]);
  out[b * 256 + o] = acc;
}
// bias2[b,o] = sum_c s1[b,c]*res3_w[o,c] + s2[b,c]*res3_w[o,256+c]
__global__ void bias2_kernel(const float* __restrict__ s1, const float* __restrict__ s2,
                             const float* __restrict__ w, float* __restrict__ out) {
  int b = blockIdx.x, o = threadIdx.x;
  const float* wr = w + (size_t)o * 768;
  const float* a = s1 + b * 256;
  const float* c2 = s2 + b * 256;
  float acc = 0.0f;
  for (int c = 0; c < 256; c++) acc += a[c] * wr[c] + c2[c] * wr[256 + c];
  out[b * 256 + o] = acc;
}
// probias[b,o] = sum_c pro[b,c]*w_proj_x[o,256+c]
__global__ void probias_kernel(const float* __restrict__ pro, const float* __restrict__ w,
                               float* __restrict__ out) {
  int b = blockIdx.x, o = threadIdx.x;
  const float* wr = w + (size_t)o * 512;
  const float* p = pro + b * 256;
  float acc = 0.0f;
  for (int c = 0; c < 256; c++) acc += p[c] * wr[256 + c];
  out[b * 256 + o] = acc;
}

// classifier head + pseudo-mask agreement masks
__global__ void cls_pred_kernel(const float* __restrict__ clsT, const float* __restrict__ w2,
                                const float* __restrict__ b2, const float* __restrict__ mask,
                                float* __restrict__ m1, float* __restrict__ m2) {
  int m = blockIdx.x * 256 + threadIdx.x;
  const float* f = clsT + (size_t)m * 256;
  float a0 = b2[0], a1 = b2[1];
  for (int c = 0; c < 256; c++) {
    float v = f[c];
    a0 += v * w2[c];
    a1 += v * w2[256 + c];
  }
  float p = (a1 > a0) ? 1.0f : 0.0f;
  float mk = mask[m];
  m1[m] = (mk != 0.0f && p == 1.0f) ? 1.0f : 0.0f;
  m2[m] = (mk != 0.0f && p == 0.0f) ? 1.0f : 0.0f;
}

__global__ void rownorm_kernel(const float* __restrict__ v, float* __restrict__ out) {
  int b = blockIdx.x;
  float t = v[b * 256 + threadIdx.x];
  t = blockRedSum(t * t);
  if (threadIdx.x == 0) out[b] = sqrtf(t);
}

// cosine scores of k rows vs fg/bg prototypes (one wave per row)
__global__ __launch_bounds__(256) void score_kernel(
    const float* __restrict__ k, const float* __restrict__ fg, const float* __restrict__ bg,
    const float* __restrict__ fgn, const float* __restrict__ bgn,
    float* __restrict__ fgs, float* __restrict__ bgs) {
  int wid = threadIdx.x >> 5, lane = threadIdx.x & 31;
  int m = blockIdx.x * 8 + wid;
  int b = m >> 12;
  const float* kr = k + (size_t)m * 256;
  const float* fgb = fg + b * 256;
  const float* bgb = bg + b * 256;
  float skk = 0.0f, skf = 0.0f, skb = 0.0f;
#pragma unroll
  for (int j = 0; j < 8; j++) {
    int c = lane + 32 * j;
    float kv = kr[c];
    skk += kv * kv; skf += kv * fgb[c]; skb += kv * bgb[c];
  }
  skk = waveRedSum(skk); skf = waveRedSum(skf); skb = waveRedSum(skb);
  if (lane == 0) {
    float kn = sqrtf(skk);
    fgs[m] = skf / (kn * fgn[b] + 1e-7f);
    bgs[m] = skb / (kn * bgn[b] + 1e-7f);
  }
}

__global__ void minmax_kernel(const float* __restrict__ s, float* __restrict__ lo,
                              float* __restrict__ hi) {
  int b = blockIdx.x;
  float mn = 3.4e38f, mx = -3.4e38f;
  for (int j = threadIdx.x; j < Npix; j += 256) {
    float v = s[(size_t)b * Npix + j];
    mn = fminf(mn, v); mx = fmaxf(mx, v);
  }
  mn = blockRedMin(mn);
  mx = blockRedMax(mx);
  if (threadIdx.x == 0) { lo[b] = mn; hi[b] = mx; }
}

__global__ void scores_kernel(const float* __restrict__ fgs, const float* __restrict__ bgs,
                              const float* __restrict__ flo, const float* __restrict__ fhi,
                              const float* __restrict__ blo, const float* __restrict__ bhi,
                              float* __restrict__ scores, float* __restrict__ pseudo) {
  int m = blockIdx.x * 256 + threadIdx.x;
  int b = m >> 12;
  float f = (fgs[m] - flo[b]) / (fhi[b] - flo[b] + 1e-7f);
  float g = (bgs[m] - blo[b]) / (bhi[b] - blo[b] + 1e-7f);
  float sc = f - g;
  scores[m] = sc;
  pseudo[m] = sc;
}

__global__ void softmax_kernel(const float* __restrict__ scores, float* __restrict__ probs) {
  int b = blockIdx.x;
  float mx = -3.4e38f;
  for (int j = threadIdx.x; j < Npix; j += 256) {
    float v = scores[(size_t)b * Npix + j];
    v += (v < 0.0f) ? -100.0f : 0.0f;
    mx = fmaxf(mx, v);
  }
  mx = blockRedMax(mx);
  float se = 0.0f;
  for (int j = threadIdx.x; j < Npix; j += 256) {
    float v = scores[(size_t)b * Npix + j];
    v += (v < 0.0f) ? -100.0f : 0.0f;
    se += expf(v - mx);
  }
  se = blockRedSum(se);
  float inv = 1.0f / se;
  for (int j = threadIdx.x; j < Npix; j += 256) {
    float v = scores[(size_t)b * Npix + j];
    v += (v < 0.0f) ? -100.0f : 0.0f;
    probs[(size_t)b * Npix + j] = expf(v - mx) * inv;
  }
}

__global__ void qpro_kernel(const float* __restrict__ probs, const float* __restrict__ v,
                            float* __restrict__ qp) {
  int b = blockIdx.x, c = threadIdx.x;
  const float* pb = probs + (size_t)b * Npix;
  const float* vb = v + (size_t)b * Npix * 256;
  float acc = 0.0f;
  for (int n = 0; n < Npix; n++) acc += pb[n] * vb[(size_t)n * 256 + c];
  qp[b * 256 + c] = acc;
}

__global__ void pro_kernel(const float* __restrict__ qp, const float* __restrict__ fg,
                           const float* __restrict__ fgn, float* __restrict__ pro) {
  int b = blockIdx.x, c = threadIdx.x;
  float q = qp[b * 256 + c], f = fg[b * 256 + c];
  float dot = blockRedSum(q * f);
  float qn2 = blockRedSum(q * q);
  float sim = dot / (sqrtf(qn2) * fgn[b] + 1e-7f);
  sim = (sim + 1.0f) * 0.5f;
  pro[b * 256 + c] = sim * f + (1.0f - sim) * q;
}

// ---------------------------------------------------------------------------
extern "C" void kernel_launch(void* const* d_in, const int* in_sizes, int n_in,
                              void* d_out, int out_size, void* d_ws, size_t ws_size,
                              hipStream_t stream) {
  (void)in_sizes; (void)n_in; (void)out_size; (void)ws_size;

  const float* x        = (const float*)d_in[0];
  const float* y        = (const float*)d_in[1];
  const float* mask     = (const float*)d_in[2];
  const float* ln1_g    = (const float*)d_in[3];
  const float* ln1_b    = (const float*)d_in[4];
  const float* wq       = (const float*)d_in[5];
  const float* wk       = (const float*)d_in[6];
  const float* wv       = (const float*)d_in[7];
  const float* w_proj_x = (const float*)d_in[8];
  const float* res1_w   = (const float*)d_in[9];
  const float* res2_w1  = (const float*)d_in[10];
  const float* res2_w2  = (const float*)d_in[11];
  const float* res3_w   = (const float*)d_in[12];
  const float* cls_w1   = (const float*)d_in[13];
  const float* cls_w2   = (const float*)d_in[14];
  const float* cls_b2   = (const float*)d_in[15];
  const float* ln2_g    = (const float*)d_in[16];
  const float* ln2_b    = (const float*)d_in[17];
  const float* fc1_w    = (const float*)d_in[18];
  const float* fc1_b    = (const float*)d_in[19];
  const float* fc2_w    = (const float*)d_in[20];
  const float* fc2_b    = (const float*)d_in[21];

  float* out_xo = (float*)d_out;
  float* out_pm = out_xo + (size_t)Mrow * Cch;

  // ---- workspace carving (requires ~410 MB) ----
  char* ws = (char*)d_ws;
  size_t off = 0;
  auto carve = [&](size_t bytes) -> void* {
    void* p = ws + off;
    off += (bytes + 255) & ~(size_t)255;
    return p;
  };
  const size_t SLOT = (size_t)Mrow * Cch * 4;  // 64 MiB
  float* q    = (float*)carve(SLOT);
  float* kb   = (float*)carve(SLOT);
  float* vb   = (float*)carve(SLOT);
  float* buf3 = (float*)carve(SLOT);  // f1 -> clsT -> h2 ; also xn/yn bf16 early
  float* buf4 = (float*)carve(SLOT);  // t -> q_xiu -> xo_ln(bf16)
  float* buf5 = (float*)carve(SLOT);  // f1r -> xo
  __bf16* xn_bf  = (__bf16*)buf3;                    // M*C bf16 = half slot
  __bf16* yn_bf  = (__bf16*)((char*)buf3 + SLOT / 2);
  __bf16* xo_ln  = (__bf16*)buf4;

  __bf16* wq_bf   = (__bf16*)carve(Cch * Cch * 2);
  __bf16* wk_bf   = (__bf16*)carve(Cch * Cch * 2);
  __bf16* wv_bf   = (__bf16*)carve(Cch * Cch * 2);
  __bf16* res1a   = (__bf16*)carve(Cch * Cch * 2);
  __bf16* res3c   = (__bf16*)carve(Cch * Cch * 2);
  __bf16* wp1_bf  = (__bf16*)carve(Cch * Cch * 2);
  __bf16* fc1_bf  = (__bf16*)carve(Cch * Cch * 2);
  __bf16* fc2_bf  = (__bf16*)carve(Cch * Cch * 2);
  __bf16* w1t     = (__bf16*)carve((size_t)9 * Cch * Cch * 2);
  __bf16* w2t     = (__bf16*)carve((size_t)9 * Cch * Cch * 2);
  __bf16* clst    = (__bf16*)carve((size_t)9 * Cch * Cch * 2);

  float* fgpro   = (float*)carve(Bsz * Cch * 4);
  float* bias1   = (float*)carve(Bsz * Cch * 4);
  float* s1      = (float*)carve(Bsz * Cch * 4);
  float* s2      = (float*)carve(Bsz * Cch * 4);
  float* bias2   = (float*)carve(Bsz * Cch * 4);
  float* fg      = (float*)carve(Bsz * Cch * 4);
  float* bg      = (float*)carve(Bsz * Cch * 4);
  float* qp      = (float*)carve(Bsz * Cch * 4);
  float* pro     = (float*)carve(Bsz * Cch * 4);
  float* probias = (float*)carve(Bsz * Cch * 4);
  float* areaM   = (float*)carve(Bsz * 4);
  float* area1   = (float*)carve(Bsz * 4);
  float* area2   = (float*)carve(Bsz * 4);
  float* areaI   = (float*)carve(Bsz * 4);
  float* fgn     = (float*)carve(Bsz * 4);
  float* bgn     = (float*)carve(Bsz * 4);
  float* flo     = (float*)carve(Bsz * 4);
  float* fhi     = (float*)carve(Bsz * 4);
  float* blo     = (float*)carve(Bsz * 4);
  float* bhi     = (float*)carve(Bsz * 4);
  float* m1v     = (float*)carve((size_t)Mrow * 4);
  float* m2v     = (float*)carve((size_t)Mrow * 4);
  float* fgs     = (float*)carve((size_t)Mrow * 4);
  float* bgs     = (float*)carve((size_t)Mrow * 4);
  float* scoresb = (float*)carve((size_t)Mrow * 4);
  float* probs   = (float*)carve((size_t)Mrow * 4);

  const dim3 gGemm(Mrow / 128, 2);
  const int cvtBlocks = (Cch * Cch) / 256;          // 256
  const int prepBlocks = (9 * Cch * Cch) / 256;     // 2304

  // ---- weight conversion / rearrangement ----
  cvt_rows_kernel<<<cvtBlocks, 256, 0, stream>>>(wq, wq_bf, 256, 0);
  cvt_rows_kernel<<<cvtBlocks, 256, 0, stream>>>(wk, wk_bf, 256, 0);
  cvt_rows_kernel<<<cvtBlocks, 256, 0, stream>>>(wv, wv_bf, 256, 0);
  cvt_rows_kernel<<<cvtBlocks, 256, 0, stream>>>(res1_w, res1a, 768, 0);
  cvt_rows_kernel<<<cvtBlocks, 256, 0, stream>>>(res3_w, res3c, 768, 512);
  cvt_rows_kernel<<<cvtBlocks, 256, 0, stream>>>(w_proj_x, wp1_bf, 512, 0);
  cvt_rows_kernel<<<cvtBlocks, 256, 0, stream>>>(fc1_w, fc1_bf, 256, 0);
  cvt_rows_kernel<<<cvtBlocks, 256, 0, stream>>>(fc2_w, fc2_bf, 256, 0);
  conv_prep_kernel<<<prepBlocks, 256, 0, stream>>>(res2_w1, w1t);
  conv_prep_kernel<<<prepBlocks, 256, 0, stream>>>(res2_w2, w2t);
  conv_prep_kernel<<<prepBlocks, 256, 0, stream>>>(cls_w1, clst);

  // ---- LN1 -> bf16 ----
  ln_kernel<<<Mrow / 8, 256, 0, stream>>>(x, ln1_g, ln1_b, xn_bf);
  ln_kernel<<<Mrow / 8, 256, 0, stream>>>(y, ln1_g, ln1_b, yn_bf);

  // ---- q/k/v projections (WMMA) ----
  gemm256_kernel<0, false, false, false, true><<<gGemm, 256, 0, stream>>>(yn_bf, wq_bf, q,  nullptr, nullptr, nullptr);
  gemm256_kernel<0, false, false, false, true><<<gGemm, 256, 0, stream>>>(xn_bf, wk_bf, kb, nullptr, nullptr, nullptr);
  gemm256_kernel<0, false, false, false, true><<<gGemm, 256, 0, stream>>>(xn_bf, wv_bf, vb, nullptr, nullptr, nullptr);

  // ---- prototype pooling, res1 (concat folded into per-batch bias) ----
  area_kernel<<<Bsz, 256, 0, stream>>>(mask, areaM, 0);
  wgap_kernel<<<Bsz, 256, 0, stream>>>(q, mask, 0, areaM, fgpro);
  bias1_kernel<<<Bsz, 256, 0, stream>>>(fgpro, res1_w, bias1);
  gemm256_kernel<1, true, false, false, false><<<gGemm, 256, 0, stream>>>(q, res1a, buf3, bias1, nullptr, nullptr);

  // ---- residual conv block + classifier head ----
  conv3x3_kernel<false><<<gGemm, 256, 0, stream>>>(buf3, w1t, buf4, nullptr);
  conv3x3_kernel<true ><<<gGemm, 256, 0, stream>>>(buf4, w2t, buf5, buf3);
  conv3x3_kernel<false><<<gGemm, 256, 0, stream>>>(buf5, clst, buf3, nullptr);
  cls_pred_kernel<<<Mrow / 256, 256, 0, stream>>>(buf3, cls_w2, cls_b2, mask, m1v, m2v);

  // ---- self-adaptive GAPs + res3 (concat folded into bias) ----
  area_kernel<<<Bsz, 256, 0, stream>>>(m1v, area1, 0);
  area_kernel<<<Bsz, 256, 0, stream>>>(m2v, area2, 0);
  wgap_kernel<<<Bsz, 256, 0, stream>>>(q, m1v, 0, area1, s1);
  wgap_kernel<<<Bsz, 256, 0, stream>>>(q, m2v, 0, area2, s2);
  bias2_kernel<<<Bsz, 256, 0, stream>>>(s1, s2, res3_w, bias2);
  gemm256_kernel<1, true, false, false, false><<<gGemm, 256, 0, stream>>>(q, res3c, buf4, bias2, nullptr, nullptr);

  // ---- fg/bg prototypes + cosine scores ----
  area_kernel<<<Bsz, 256, 0, stream>>>(mask, areaI, 1);
  wgap_kernel<<<Bsz, 256, 0, stream>>>(buf4, mask, 0, areaM, fg);
  wgap_kernel<<<Bsz, 256, 0, stream>>>(buf4, mask, 1, areaI, bg);
  rownorm_kernel<<<Bsz, 256, 0, stream>>>(fg, fgn);
  rownorm_kernel<<<Bsz, 256, 0, stream>>>(bg, bgn);
  score_kernel<<<Mrow / 8, 256, 0, stream>>>(kb, fg, bg, fgn, bgn, fgs, bgs);
  minmax_kernel<<<Bsz, 256, 0, stream>>>(fgs, flo, fhi);
  minmax_kernel<<<Bsz, 256, 0, stream>>>(bgs, blo, bhi);
  scores_kernel<<<Mrow / 256, 256, 0, stream>>>(fgs, bgs, flo, fhi, blo, bhi, scoresb, out_pm);

  // ---- attention pooling + prototype mixing ----
  softmax_kernel<<<Bsz, 256, 0, stream>>>(scoresb, probs);
  qpro_kernel<<<Bsz, 256, 0, stream>>>(probs, vb, qp);
  pro_kernel<<<Bsz, 256, 0, stream>>>(qp, fg, fgn, pro);
  probias_kernel<<<Bsz, 256, 0, stream>>>(pro, w_proj_x, probias);

  // ---- projection residual + MLP ----
  gemm256_kernel<0, true, false, true, false><<<gGemm, 256, 0, stream>>>(x, wp1_bf, buf5, probias, nullptr, x);
  ln_kernel<<<Mrow / 8, 256, 0, stream>>>(buf5, ln2_g, ln2_b, xo_ln);
  gemm256_kernel<2, false, true, false, true><<<gGemm, 256, 0, stream>>>(xo_ln, fc1_bf, buf3, nullptr, fc1_b, nullptr);
  gemm256_kernel<0, false, true, true, false><<<gGemm, 256, 0, stream>>>(buf3, fc2_bf, out_xo, nullptr, fc2_b, buf5);
}
